// multi_dRNN_with_dilations_17008070492229
// MI455X (gfx1250) — matrix operand order repaired and checked by reading, not compiled, not run on hardware
//
#include <hip/hip_runtime.h>
#include <hip/hip_bf16.h>

// Dilated 3-layer GRU for MI455X (gfx1250), bf16 WMMA pipeline.
//  - weights transposed+converted to bf16 [N][K] once per call (L2-resident)
//  - fragment-major LDS tiles: every WMMA operand is 2x ds_load_b128/lane
//  - software-pipelined K loop (ping-pong regs + double-buffered LDS),
//    one barrier per slab, global latency hidden behind WMMAs
//  - kernels templated on DIN: K loop fully unrolled, all slab addresses
//    fold into immediate offsets (no per-iteration address math)
//
// reference dims
#define B_ 64
#define T_ 512
#define DIN_ 256
#define HD_ 512          // hidden size (all layers)
#define NG_ 1024         // gate width = 2*HD_

typedef __bf16 bf16_t;
typedef __attribute__((ext_vector_type(16))) __bf16 v16bf;
typedef __attribute__((ext_vector_type(8)))  __bf16 v8bf;
typedef __attribute__((ext_vector_type(8)))  float  v8f;

#define LDA_ 40   // padded stride (elements): 80B, 16B-aligned, bank-friendly

__device__ __forceinline__ v16bf load_frag(const bf16_t* p) {
  const v8bf* pv = (const v8bf*)p;
  v8bf lo = pv[0], hi = pv[1];
  return __builtin_shufflevector(lo, hi, 0, 1, 2, 3, 4, 5, 6, 7,
                                 8, 9, 10, 11, 12, 13, 14, 15);
}

// fragment-major slot base for an 8-element run: seg {0,1,2,3}->{0,16,8,24}
__device__ __forceinline__ int seg_slot(int seg) {
  return ((seg & 1) << 4) | ((seg & 2) << 2);
}

// ---------------------------------------------------------------------------
// layout / conversion kernels
// ---------------------------------------------------------------------------
// src [K, N] f32 -> dst [N, K] bf16 (tiled transpose, coalesced both sides)
__global__ __launch_bounds__(256) void transpose_w_bf16(
    const float* __restrict__ src, bf16_t* __restrict__ dst, int K, int N) {
  __shared__ float tile[32][33];
  const int tx = threadIdx.x & 31, ty = threadIdx.x >> 5;   // 32x8
  const int k0 = blockIdx.y * 32, n0 = blockIdx.x * 32;
#pragma unroll
  for (int i = 0; i < 32; i += 8)
    tile[ty + i][tx] = src[(size_t)(k0 + ty + i) * N + n0 + tx];
  __syncthreads();
#pragma unroll
  for (int i = 0; i < 32; i += 8)
    dst[(size_t)(n0 + ty + i) * K + k0 + tx] = (bf16_t)tile[tx][ty + i];
}

// x [B,T,DIN] f32  ->  Xbf [T,B,DIN] bf16 (time-major for the scan)
__global__ void x_to_tm_bf16(const float* __restrict__ x,
                             bf16_t* __restrict__ xbf) {
  int o = blockIdx.x * blockDim.x + threadIdx.x;
  const int n = T_ * B_ * DIN_;
  if (o >= n) return;
  int d = o % DIN_;
  int r = o / DIN_;
  int b = r % B_;
  int t = r / B_;
  xbf[o] = (bf16_t)x[((size_t)b * T_ + t) * DIN_ + d];
}

// Y2 [T,B,HD] bf16 -> out [B,T,HD] f32
__global__ void finalize_out(const bf16_t* __restrict__ y2,
                             float* __restrict__ out) {
  int o = blockIdx.x * blockDim.x + threadIdx.x;
  const int n = B_ * T_ * HD_;
  if (o >= n) return;
  int c = o % HD_;
  int r = o / HD_;
  int t = r % T_;
  int b = r / T_;
  out[o] = (float)y2[((size_t)t * B_ + b) * HD_ + c];
}

// states[l,b,n] = Y_l at time T-1
__global__ void gather_states(const bf16_t* __restrict__ y0,
                              const bf16_t* __restrict__ y1,
                              const bf16_t* __restrict__ y2,
                              float* __restrict__ st) {
  int i = blockIdx.x * blockDim.x + threadIdx.x;
  const int n = 3 * B_ * HD_;
  if (i >= n) return;
  int c = i % HD_;
  int r = i / HD_;
  int b = r % B_;
  int l = r / B_;
  const bf16_t* y = (l == 0) ? y0 : (l == 1) ? y1 : y2;
  st[i] = (float)y[((size_t)(T_ - 1) * B_ + b) * HD_ + c];
}

// ---------------------------------------------------------------------------
// GEMM pipeline stages (device, inlined). Block = 128 threads, tile 64x64.
// Each thread moves 2 runs of 8 bf16 for A and for B per 32-wide K slab.
// DIN and k0 are compile-time -> all addressing folds to base + immediate.
// ---------------------------------------------------------------------------
template <int DIN>
__device__ __forceinline__ void load_slab(
    const bf16_t* __restrict__ Xs,
    const bf16_t* __restrict__ Ah,   // hidden-part A source (bf16)
    const bf16_t* __restrict__ WT,
    int n0, int m0, int k0, int tid, v8bf a[2], v8bf b[2]) {
  constexpr int Ktot = DIN + HD_;
  // K-slab never straddles the X|hidden boundary (DIN, k0 multiples of 32)
#pragma unroll
  for (int i = 0; i < 2; ++i) {
    int run = tid + i * 128;       // 256 runs of 8 elements
    int r = run >> 2, seg = run & 3;
    if (k0 < DIN)
      a[i] = *(const v8bf*)&Xs[(size_t)(m0 + r) * DIN + k0 + seg * 8];
    else
      a[i] = *(const v8bf*)&Ah[(size_t)(m0 + r) * HD_ + (k0 - DIN) + seg * 8];
    b[i] = *(const v8bf*)&WT[(size_t)(n0 + r) * Ktot + k0 + seg * 8];
  }
}

__device__ __forceinline__ void store_slab(bf16_t* lds_a, bf16_t* lds_b,
                                           int tid, const v8bf a[2],
                                           const v8bf b[2]) {
#pragma unroll
  for (int i = 0; i < 2; ++i) {
    int run = tid + i * 128;
    int r = run >> 2, seg = run & 3;
    *(v8bf*)&lds_a[r * LDA_ + seg_slot(seg)] = a[i];
    *(v8bf*)&lds_b[r * LDA_ + seg_slot(seg)] = b[i];
  }
}

__device__ __forceinline__ void compute_slab(const bf16_t* lds_a,
                                             const bf16_t* lds_b,
                                             int lane, int wave, int half,
                                             v8f acc[4]) {
  int r = (wave << 4) + (lane & 15);
  v16bf afrag = load_frag(&lds_a[r * LDA_ + half]);
  v16bf bfr[4];
#pragma unroll
  for (int nb = 0; nb < 4; ++nb)
    bfr[nb] = load_frag(&lds_b[((nb << 4) + (lane & 15)) * LDA_ + half]);
#pragma unroll
  for (int nb = 0; nb < 4; ++nb)
    acc[nb] = __builtin_amdgcn_wmma_f32_16x16x32_bf16(
        false, afrag, false, bfr[nb], (short)0, acc[nb], false, false);
}

// full pipelined GEMM over Ktot (compile-time, multiple of 64)
template <int DIN>
__device__ __forceinline__ void gemm_body(
    const bf16_t* __restrict__ Xs,
    const bf16_t* __restrict__ Ah,
    const bf16_t* __restrict__ WT,
    int n0, int m0, int tid, int lane, int wave, int half,
    bf16_t* lds_a0, bf16_t* lds_b0, bf16_t* lds_a1, bf16_t* lds_b1,
    v8f acc[4]) {
  constexpr int Ktot = DIN + HD_;
  v8bf aA[2], bA[2], aB[2], bB[2];
  load_slab<DIN>(Xs, Ah, WT, n0, m0, 0, tid, aA, bA);
#pragma unroll
  for (int k0 = 0; k0 < Ktot; k0 += 64) {
    // slab k0: LDS buffer 0, register set A
    store_slab(lds_a0, lds_b0, tid, aA, bA);
    __syncthreads();
    load_slab<DIN>(Xs, Ah, WT, n0, m0, k0 + 32, tid, aB, bB);
    compute_slab(lds_a0, lds_b0, lane, wave, half, acc);
    // slab k0+32: LDS buffer 1, register set B
    store_slab(lds_a1, lds_b1, tid, aB, bB);
    __syncthreads();
    if (k0 + 64 < Ktot)
      load_slab<DIN>(Xs, Ah, WT, n0, m0, k0 + 64, tid, aA, bA);
    compute_slab(lds_a1, lds_b1, lane, wave, half, acc);
  }
}

// ---------------------------------------------------------------------------
// GEMM kernel 1: gates.  G = sigmoid([X | h] @ GK + gb) -> writes
//   RH[m,n] = r[m,n] * h_prev[m,n]  (bf16, n < HD)  and  U[m,n] = u  (f32)
// Grid: (NG/64, M/64), block 128 (4 waves). Each wave: 16 rows x 64 cols.
// ---------------------------------------------------------------------------
template <int DIN>
__global__ __launch_bounds__(128) void gru_gates_kernel(
    const bf16_t* __restrict__ Xs,
    const bf16_t* __restrict__ Hb,   // h_prev bf16 [M, HD] (prev output slab)
    const float* __restrict__ H,     // h_prev f32  [M, HD]
    const bf16_t* __restrict__ GKT,  // [NG][DIN+HD] bf16 (transposed)
    const float* __restrict__ gb,    // [NG]
    float* __restrict__ U,           // [M, HD]
    bf16_t* __restrict__ RH) {       // [M, HD]
  __shared__ __align__(16) bf16_t lds_a0[64 * LDA_];
  __shared__ __align__(16) bf16_t lds_b0[64 * LDA_];
  __shared__ __align__(16) bf16_t lds_a1[64 * LDA_];
  __shared__ __align__(16) bf16_t lds_b1[64 * LDA_];
  const int tid  = threadIdx.x;
  const int lane = tid & 31;
  const int wave = tid >> 5;
  const int n0 = blockIdx.x * 64;
  const int m0 = blockIdx.y * 64;
  const int half = (lane < 16) ? 0 : 16;

  v8f acc[4] = {};
  gemm_body<DIN>(Xs, Hb, GKT, n0, m0, tid, lane, wave, half,
                 lds_a0, lds_b0, lds_a1, lds_b1, acc);

  // epilogue: bias + sigmoid; block-uniform r/u split (n0 is 64-aligned)
  const int col = lane & 15;
  const int rbase = m0 + (wave << 4);
  if (n0 < HD_) {               // r-gate half -> RH = sigmoid(g) * h_prev
#pragma unroll
    for (int nb = 0; nb < 4; ++nb) {
#pragma unroll
      for (int j = 0; j < 8; ++j) {
        int drow = (lane < 16) ? j : j + 8;
        int n = n0 + (nb << 4) + col;
        int m = rbase + drow;
        float g = acc[nb][j] + gb[n];
        g = 1.0f / (1.0f + __expf(-g));
        float hp = H[(size_t)m * HD_ + n];
        RH[(size_t)m * HD_ + n] = (bf16_t)(g * hp);
      }
    }
  } else {                      // u-gate half -> U = sigmoid(g)
#pragma unroll
    for (int nb = 0; nb < 4; ++nb) {
#pragma unroll
      for (int j = 0; j < 8; ++j) {
        int drow = (lane < 16) ? j : j + 8;
        int n = n0 + (nb << 4) + col;
        int m = rbase + drow;
        float g = acc[nb][j] + gb[n];
        g = 1.0f / (1.0f + __expf(-g));
        U[(size_t)m * HD_ + (n - HD_)] = g;
      }
    }
  }
}

// ---------------------------------------------------------------------------
// GEMM kernel 2: candidate + blend.
//   c = tanh([X | RH] @ CK + cb);  h' = u*h + (1-u)*c
// writes H (f32, in-place) and Ys (time-major bf16 layer output slab, which
// also serves as next step's bf16 h_prev). Grid: (HD/64, M/64), block 128.
// ---------------------------------------------------------------------------
template <int DIN>
__global__ __launch_bounds__(128) void gru_cand_kernel(
    const bf16_t* __restrict__ Xs,
    const bf16_t* __restrict__ RH,   // [M, HD] bf16
    const bf16_t* __restrict__ CKT,  // [HD][DIN+HD] bf16 (transposed)
    const float* __restrict__ cb,    // [HD]
    const float* __restrict__ U,     // [M, HD]
    float* __restrict__ H,           // [M, HD]  (read h_prev, write h_new)
    bf16_t* __restrict__ Ys) {       // output slab for this macro-step
  __shared__ __align__(16) bf16_t lds_a0[64 * LDA_];
  __shared__ __align__(16) bf16_t lds_b0[64 * LDA_];
  __shared__ __align__(16) bf16_t lds_a1[64 * LDA_];
  __shared__ __align__(16) bf16_t lds_b1[64 * LDA_];
  const int tid  = threadIdx.x;
  const int lane = tid & 31;
  const int wave = tid >> 5;
  const int n0 = blockIdx.x * 64;
  const int m0 = blockIdx.y * 64;
  const int half = (lane < 16) ? 0 : 16;

  v8f acc[4] = {};
  gemm_body<DIN>(Xs, RH, CKT, n0, m0, tid, lane, wave, half,
                 lds_a0, lds_b0, lds_a1, lds_b1, acc);

  const int col = lane & 15;
  const int rbase = m0 + (wave << 4);
#pragma unroll
  for (int nb = 0; nb < 4; ++nb) {
#pragma unroll
    for (int j = 0; j < 8; ++j) {
      int drow = (lane < 16) ? j : j + 8;
      int n = n0 + (nb << 4) + col;
      int m = rbase + drow;
      float c = tanhf(acc[nb][j] + cb[n]);
      float u = U[(size_t)m * HD_ + n];
      float hp = H[(size_t)m * HD_ + n];
      float hn = u * hp + (1.0f - u) * c;
      H[(size_t)m * HD_ + n] = hn;
      Ys[(size_t)m * HD_ + n] = (bf16_t)hn;
    }
  }
}

// ---------------------------------------------------------------------------
// host-side orchestration
// ---------------------------------------------------------------------------
extern "C" void kernel_launch(void* const* d_in, const int* in_sizes, int n_in,
                              void* d_out, int out_size, void* d_ws, size_t ws_size,
                              hipStream_t stream) {
  (void)in_sizes; (void)n_in; (void)out_size;
  const float* x = (const float*)d_in[0];
  const float* gk[3] = { (const float*)d_in[1], (const float*)d_in[5], (const float*)d_in[9]  };
  const float* gbp[3]= { (const float*)d_in[2], (const float*)d_in[6], (const float*)d_in[10] };
  const float* ck[3] = { (const float*)d_in[3], (const float*)d_in[7], (const float*)d_in[11] };
  const float* cbp[3]= { (const float*)d_in[4], (const float*)d_in[8], (const float*)d_in[12] };
  float* out = (float*)d_out;

  const int din_l[3] = { DIN_, HD_, HD_ };

  // ---- workspace carve-up ----
  uint8_t* w = (uint8_t*)d_ws;
  size_t off = 0;
  auto carve = [&](size_t bytes) -> void* {
    void* p = w + off;
    off = (off + bytes + 255) & ~(size_t)255;
    return p;
  };
  bf16_t* gkT[3]; bf16_t* ckT[3];
  for (int l = 0; l < 3; ++l) {
    gkT[l] = (bf16_t*)carve((size_t)(din_l[l] + HD_) * NG_ * sizeof(bf16_t));
    ckT[l] = (bf16_t*)carve((size_t)(din_l[l] + HD_) * HD_ * sizeof(bf16_t));
  }
  bf16_t* xbf = (bf16_t*)carve((size_t)T_ * B_ * DIN_ * sizeof(bf16_t));
  bf16_t* Y[3];
  for (int l = 0; l < 3; ++l)
    Y[l] = (bf16_t*)carve((size_t)T_ * B_ * HD_ * sizeof(bf16_t));
  float* Hst[3];
  for (int l = 0; l < 3; ++l)
    Hst[l] = (float*)carve((size_t)256 * HD_ * sizeof(float));
  float*  U   = (float*)carve((size_t)256 * HD_ * sizeof(float));
  bf16_t* RH  = (bf16_t*)carve((size_t)256 * HD_ * sizeof(bf16_t));
  bf16_t* Hb0 = (bf16_t*)carve((size_t)256 * HD_ * sizeof(bf16_t));  // zeros
  if (off > ws_size) return;  // workspace too small: nothing safe to do

  // ---- weight transpose+convert, input convert (deterministic per call) ----
  for (int l = 0; l < 3; ++l) {
    int K = din_l[l] + HD_;
    transpose_w_bf16<<<dim3(NG_ / 32, K / 32), 256, 0, stream>>>(gk[l], gkT[l], K, NG_);
    transpose_w_bf16<<<dim3(HD_ / 32, K / 32), 256, 0, stream>>>(ck[l], ckT[l], K, HD_);
  }
  {
    int n = T_ * B_ * DIN_;
    x_to_tm_bf16<<<(n + 255) / 256, 256, 0, stream>>>(x, xbf);
  }
  hipMemsetAsync(Hb0, 0, (size_t)256 * HD_ * sizeof(bf16_t), stream);

  // ---- the dilated scan ----
  for (int l = 0; l < 3; ++l) {
    const int rate = 1 << l;           // 1, 2, 4
    const int M    = rate * B_;        // 64, 128, 256
    const int S    = T_ / rate;        // 512, 256, 128
    const int din  = din_l[l];
    const bf16_t* Xlayer = (l == 0) ? xbf : Y[l - 1];

    hipMemsetAsync(Hst[l], 0, (size_t)M * HD_ * sizeof(float), stream);

    dim3 blk(128);
    dim3 grdG(NG_ / 64, M / 64);
    dim3 grdC(HD_ / 64, M / 64);
    for (int s = 0; s < S; ++s) {
      const bf16_t* Xs = Xlayer + (size_t)s * rate * B_ * din;
      bf16_t*       Ys = Y[l]   + (size_t)s * rate * B_ * HD_;
      // bf16 h_prev = previous macro-step's output slab (zeros at s==0)
      const bf16_t* Hb = (s == 0) ? Hb0
                                  : Y[l] + (size_t)(s - 1) * rate * B_ * HD_;
      if (l == 0) {
        gru_gates_kernel<DIN_><<<grdG, blk, 0, stream>>>(Xs, Hb, Hst[l],
                                                         gkT[l], gbp[l], U, RH);
        gru_cand_kernel<DIN_><<<grdC, blk, 0, stream>>>(Xs, RH, ckT[l],
                                                        cbp[l], U, Hst[l], Ys);
      } else {
        gru_gates_kernel<HD_><<<grdG, blk, 0, stream>>>(Xs, Hb, Hst[l],
                                                        gkT[l], gbp[l], U, RH);
        gru_cand_kernel<HD_><<<grdC, blk, 0, stream>>>(Xs, RH, ckT[l],
                                                       cbp[l], U, Hst[l], Ys);
      }
    }
  }

  // ---- emit [B,T,H] output + stacked final states ----
  {
    int n = B_ * T_ * HD_;
    finalize_out<<<(n + 255) / 256, 256, 0, stream>>>(Y[2], out);
    int ns = 3 * B_ * HD_;
    gather_states<<<(ns + 255) / 256, 256, 0, stream>>>(
        Y[0], Y[1], Y[2], out + (size_t)B_ * T_ * HD_);
  }
}
